// pMHCTCR_56135222558884
// MI455X (gfx1250) — matrix operand order repaired and checked by reading, not compiled, not run on hardware
//
#include <hip/hip_runtime.h>

// ---------------------------------------------------------------------------
// MI455X (gfx1250) fused dual-MLP + paired-cosine kernel.
//
// Roofline: ~31 GFLOP vs ~404 MB of HBM traffic (17 us @ 23.3 TB/s). fp32
// VALU would be compute bound; bf16 WMMA (v_wmma_f32_16x16x32_bf16) puts the
// GEMMs on the matrix pipe and balances compute against HBM. Weights are
// repacked once into WMMA B-operand fragment layout in d_ws (44 KB, L2
// resident). Each wave processes one 16-row tile end-to-end, transposing
// C-layout -> A-layout for the second linear of each MLP through per-wave LDS.
// ---------------------------------------------------------------------------

typedef __attribute__((ext_vector_type(16))) __bf16 v16bf;
typedef __attribute__((ext_vector_type(8)))  __bf16 v8bf;
typedef __attribute__((ext_vector_type(8)))  float  v8f;

#define INV_TEMP 10.0f
#define EPS_NRM  1e-12f

// Fragment table (each fragment = 32 lanes x 16 bf16 = 512 elems = 1 KB):
//   G1: W1 30x50  -> ntiles 4, ksteps 1, frag base  0
//   G2: W2 50x70  -> ntiles 5, ksteps 2, frag base  4
//   G3: W3 70x70  -> ntiles 5, ksteps 3, frag base 14
//   G4: W4 70x70  -> ntiles 5, ksteps 3, frag base 29   (44 frags total)
#define FR1 0
#define FR2 4
#define FR3 14
#define FR4 29
#define NFRAG 44

__device__ __forceinline__ v8f vzero8() {
  v8f z = {0.f, 0.f, 0.f, 0.f, 0.f, 0.f, 0.f, 0.f};
  return z;
}

// ---------------- weight repack: row-major f32 -> bf16 B-fragment layout ----
__device__ void repack_one(const float* __restrict__ W, int K, int Nn,
                           int ksteps, int ntiles, __bf16* __restrict__ dst,
                           int tid, int nth) {
  int elems = ntiles * ksteps * 512;
  for (int idx = tid; idx < elems; idx += nth) {
    int lf    = idx >> 9;          // local fragment id
    int lane  = (idx >> 4) & 31;
    int i     = idx & 15;          // element within lane (K offset)
    int ntile = lf / ksteps;
    int kstep = lf - ntile * ksteps;
    // ISA 32x16 bf16 B layout: lanes 0-15 hold K={kbase..kbase+7, kbase+16..kbase+23}
    // for col N=lane, lanes 16-31 use kbase+8 for col N=lane-16.
    int k = kstep * 32 + ((lane & 16) ? 8 : 0) + (i < 8 ? i : i + 8);
    int n = ntile * 16 + (lane & 15);
    float v = (k < K && n < Nn) ? W[k * Nn + n] : 0.0f;
    dst[idx] = (__bf16)v;
  }
}

__global__ void repack_weights(const float* W1, const float* W2,
                               const float* W3, const float* W4,
                               __bf16* __restrict__ ws) {
  int tid = blockIdx.x * blockDim.x + threadIdx.x;
  int nth = gridDim.x * blockDim.x;
  repack_one(W1, 30, 50, 1, 4, ws + FR1 * 512, tid, nth);
  repack_one(W2, 50, 70, 2, 5, ws + FR2 * 512, tid, nth);
  repack_one(W3, 70, 70, 3, 5, ws + FR3 * 512, tid, nth);
  repack_one(W4, 70, 70, 3, 5, ws + FR4 * 512, tid, nth);
}

// ---------------- operand fragment loaders ---------------------------------
__device__ __forceinline__ v16bf bfrag(const __bf16* __restrict__ ws, int frag,
                                       int lane) {
  // 32B contiguous per lane, 32B aligned -> two global_load_b128, L2 hot.
  return *(const v16bf*)(ws + frag * 512 + lane * 16);
}

// A fragment (16x32 bf16 tile) from a row-major bf16 LDS staging buffer.
// ISA 16x32 bf16 A layout: lane L holds row M=L%16; lanes 0-15 carry
// K = {off..off+7, off+16..off+23} with off = kstep*32; lanes 16-31 use off+8.
__device__ __forceinline__ v16bf afrag_lds(const __bf16* __restrict__ h,
                                           int stride, int kstep, int lane) {
  int row = lane & 15;
  int off = kstep * 32 + ((lane & 16) ? 8 : 0);
  const __bf16* p = h + row * stride + off;
  v8bf lo = *(const v8bf*)p;         // ds_load_b128
  v8bf hi = *(const v8bf*)(p + 16);  // ds_load_b128
  return __builtin_shufflevector(lo, hi, 0, 1, 2, 3, 4, 5, 6, 7,
                                 8, 9, 10, 11, 12, 13, 14, 15);
}

// A fragment built from a row-major fp32 global row (K-tail zero padded).
__device__ __forceinline__ v16bf afrag_gmem(const float* __restrict__ rowp,
                                            int Kvalid, int kstep, int lane) {
  int off = kstep * 32 + ((lane & 16) ? 8 : 0);
  v16bf a;
#pragma unroll
  for (int i = 0; i < 8; ++i) {
    int k0 = off + i;
    int k1 = off + 16 + i;
    float f0 = (k0 < Kvalid) ? rowp[k0] : 0.0f;
    float f1 = (k1 < Kvalid) ? rowp[k1] : 0.0f;
    a[i]     = (__bf16)f0;
    a[i + 8] = (__bf16)f1;
  }
  return a;
}

__device__ __forceinline__ v8f wmma_bf16(v16bf a, v16bf b, v8f c) {
  return __builtin_amdgcn_wmma_f32_16x16x32_bf16(
      /*neg_a=*/false, a, /*neg_b=*/false, b,
      /*c_mod=*/(short)0, c, /*reuse_a=*/false, /*reuse_b=*/false);
}

// ---------------- fused per-tile kernel ------------------------------------
__global__ __launch_bounds__(256) void fused_mlp_cosine(
    const float* __restrict__ tcr, const float* __restrict__ pmhc,
    const float* __restrict__ b1, const float* __restrict__ b2,
    const float* __restrict__ b3, const float* __restrict__ b4,
    const __bf16* __restrict__ ws, float* __restrict__ out, int nrows) {
  // Per-wave C->A transpose staging (bf16). 8 waves * (2KB + 3KB) = 40 KB.
  __shared__ __attribute__((aligned(16))) __bf16 h1s[8][16 * 64];
  __shared__ __attribute__((aligned(16))) __bf16 h2s[8][16 * 96];

  const int lane = threadIdx.x & 31;
  const int wid  = threadIdx.x >> 5;
  const long long tiles = ((long long)nrows + 15) >> 4;
  const long long tile  = (long long)blockIdx.x * 8 + wid;
  if (tile >= tiles) return;  // wave-uniform: EXEC stays all-1s for WMMA

  const long long r0 = tile << 4;
  __bf16* h1 = h1s[wid];
  __bf16* h2 = h2s[wid];

  const int ln   = lane & 15;              // column-within-tile for C layout
  const int mrow = (lane & 16) ? 8 : 0;    // row-half for C layout
  long long arow = r0 + ln;                // A-operand row for this lane
  if (arow >= nrows) arow = nrows - 1;     // tail clamp (loads only)

  // Zero h2 columns 80..95 (K-pad for GEMM4; never written by GEMM3).
  for (int e = lane; e < 256; e += 32) {
    h2[(e >> 4) * 96 + 80 + (e & 15)] = (__bf16)0.0f;
  }

  // ---- GEMM3: relu(tcr[16x70] @ W3 + b3) -> h2 (16x96 bf16, padded cols=0)
  {
    const float* rp = tcr + arow * 70;
    v16bf a0 = afrag_gmem(rp, 70, 0, lane);
    v16bf a1 = afrag_gmem(rp, 70, 1, lane);
    v16bf a2 = afrag_gmem(rp, 70, 2, lane);
#pragma unroll
    for (int t = 0; t < 5; ++t) {
      v8f acc = vzero8();
      acc = wmma_bf16(a0, bfrag(ws, FR3 + t * 3 + 0, lane), acc);
      acc = wmma_bf16(a1, bfrag(ws, FR3 + t * 3 + 1, lane), acc);
      acc = wmma_bf16(a2, bfrag(ws, FR3 + t * 3 + 2, lane), acc);
      int col = t * 16 + ln;
      float bb = (col < 70) ? b3[col] : 0.0f;
#pragma unroll
      for (int r = 0; r < 8; ++r) {
        float v = fmaxf(acc[r] + bb, 0.0f);
        h2[(r + mrow) * 96 + col] = (__bf16)v;
      }
    }
  }

  // ---- GEMM4: h2 @ W4 + b4 -> Ztcr tiles (kept in registers, no relu)
  v8f zt[5];
  {
    v16bf a0 = afrag_lds(h2, 96, 0, lane);
    v16bf a1 = afrag_lds(h2, 96, 1, lane);
    v16bf a2 = afrag_lds(h2, 96, 2, lane);
#pragma unroll
    for (int t = 0; t < 5; ++t) {
      v8f acc = vzero8();
      acc = wmma_bf16(a0, bfrag(ws, FR4 + t * 3 + 0, lane), acc);
      acc = wmma_bf16(a1, bfrag(ws, FR4 + t * 3 + 1, lane), acc);
      acc = wmma_bf16(a2, bfrag(ws, FR4 + t * 3 + 2, lane), acc);
      int col = t * 16 + ln;
      float bb = (col < 70) ? b4[col] : 0.0f;
#pragma unroll
      for (int r = 0; r < 8; ++r) zt[t][r] = acc[r] + bb;
    }
  }

  // ---- GEMM1: relu(pmhc[16x30] @ W1 + b1) -> h1 (16x64 bf16)
  {
    const float* rp = pmhc + arow * 30;
    v16bf a0 = afrag_gmem(rp, 30, 0, lane);
#pragma unroll
    for (int t = 0; t < 4; ++t) {
      v8f acc = wmma_bf16(a0, bfrag(ws, FR1 + t, lane), vzero8());
      int col = t * 16 + ln;
      float bb = (col < 50) ? b1[col] : 0.0f;
#pragma unroll
      for (int r = 0; r < 8; ++r) {
        float v = fmaxf(acc[r] + bb, 0.0f);
        h1[(r + mrow) * 64 + col] = (__bf16)v;
      }
    }
  }

  // ---- GEMM2: h1 @ W2 + b2 -> Zpmhc tiles, fused with dot/norm partials
  v8f dotp = vzero8(), npq = vzero8(), ntq = vzero8();
  {
    v16bf a0 = afrag_lds(h1, 64, 0, lane);
    v16bf a1 = afrag_lds(h1, 64, 1, lane);
#pragma unroll
    for (int t = 0; t < 5; ++t) {
      v8f acc = vzero8();
      acc = wmma_bf16(a0, bfrag(ws, FR2 + t * 2 + 0, lane), acc);
      acc = wmma_bf16(a1, bfrag(ws, FR2 + t * 2 + 1, lane), acc);
      int col = t * 16 + ln;
      float bb = (col < 70) ? b2[col] : 0.0f;
#pragma unroll
      for (int r = 0; r < 8; ++r) {
        float zp = acc[r] + bb;   // padded cols give exactly 0 -> no effect
        float z  = zt[t][r];
        dotp[r] += zp * z;
        npq[r]  += zp * zp;
        ntq[r]  += z * z;
      }
    }
  }

  // ---- butterfly reduce across the 16 columns held by each lane half
#pragma unroll
  for (int r = 0; r < 8; ++r) {
    float d = dotp[r], a = npq[r], b = ntq[r];
    d += __shfl_xor(d, 8); a += __shfl_xor(a, 8); b += __shfl_xor(b, 8);
    d += __shfl_xor(d, 4); a += __shfl_xor(a, 4); b += __shfl_xor(b, 4);
    d += __shfl_xor(d, 2); a += __shfl_xor(a, 2); b += __shfl_xor(b, 2);
    d += __shfl_xor(d, 1); a += __shfl_xor(a, 1); b += __shfl_xor(b, 1);
    dotp[r] = d; npq[r] = a; ntq[r] = b;
  }

  // ---- lanes 0 / 16 write rows r0+0..7 / r0+8..15
  if (ln == 0) {
#pragma unroll
    for (int r = 0; r < 8; ++r) {
      long long orow = r0 + mrow + r;
      if (orow < nrows) {
        float denom = fmaxf(sqrtf(npq[r]), EPS_NRM) *
                      fmaxf(sqrtf(ntq[r]), EPS_NRM);
        out[orow] = dotp[r] / denom * INV_TEMP;
      }
    }
  }
}

// ---------------------------------------------------------------------------
extern "C" void kernel_launch(void* const* d_in, const int* in_sizes, int n_in,
                              void* d_out, int out_size, void* d_ws,
                              size_t ws_size, hipStream_t stream) {
  const float* tcr  = (const float*)d_in[0];
  const float* pmhc = (const float*)d_in[1];
  const float* W1 = (const float*)d_in[2];
  const float* b1 = (const float*)d_in[3];
  const float* W2 = (const float*)d_in[4];
  const float* b2 = (const float*)d_in[5];
  const float* W3 = (const float*)d_in[6];
  const float* b3 = (const float*)d_in[7];
  const float* W4 = (const float*)d_in[8];
  const float* b4 = (const float*)d_in[9];

  const int nrows = in_sizes[0] / 70;      // tcr is [N,70]
  __bf16* ws = (__bf16*)d_ws;              // needs NFRAG*512*2 = 45056 bytes

  repack_weights<<<32, 256, 0, stream>>>(W1, W2, W3, W4, ws);

  const long long tiles  = ((long long)nrows + 15) / 16;
  const int       blocks = (int)((tiles + 7) / 8);   // 8 waves per block
  fused_mlp_cosine<<<blocks, 256, 0, stream>>>(tcr, pmhc, b1, b2, b3, b4, ws,
                                               (float*)d_out, nrows);
}
